// Router_21732534518316
// MI455X (gfx1250) — matrix operand order repaired and checked by reading, not compiled, not run on hardware
//
#include <hip/hip_runtime.h>
#include <math.h>

typedef __attribute__((ext_vector_type(16))) __bf16 v16bf;
typedef __attribute__((ext_vector_type(8)))  float  v8f;

// ---------------------------------------------------------------- converts
__global__ void cvt_bf16_kernel(const float* __restrict__ in, __bf16* __restrict__ out, int n) {
  int stride = gridDim.x * blockDim.x;
  for (int i = blockIdx.x * blockDim.x + threadIdx.x; i < n; i += stride)
    out[i] = (__bf16)in[i];
}

__global__ void zero_f32_kernel(float* p, int n) {
  int i = blockIdx.x * blockDim.x + threadIdx.x;
  if (i < n) p[i] = 0.f;
}

// ---------------------------------------------------------------- tau GEMV
__global__ void tau_kernel(const float* __restrict__ x,
                           const float* __restrict__ w_attn, const float* __restrict__ b_attn,
                           const float* __restrict__ w_know, const float* __restrict__ b_know,
                           float* __restrict__ tau_all, float* __restrict__ tau_know,
                           int R, int D) {
  const int lane = threadIdx.x & 31;
  const int wrow = (blockIdx.x * blockDim.x + threadIdx.x) >> 5;
  if (wrow >= R) return;
  const float* xr = x + (size_t)wrow * D;
  float t0 = 0.f, t1 = 0.f, t2 = 0.f, t3 = 0.f;
  for (int d = lane; d < D; d += 32) {
    float xv = xr[d];
    t0 = fmaf(xv, w_attn[d * 3 + 0], t0);
    t1 = fmaf(xv, w_attn[d * 3 + 1], t1);
    t2 = fmaf(xv, w_attn[d * 3 + 2], t2);
    t3 = fmaf(xv, w_know[d], t3);
  }
  for (int off = 16; off; off >>= 1) {
    t0 += __shfl_down(t0, off);
    t1 += __shfl_down(t1, off);
    t2 += __shfl_down(t2, off);
    t3 += __shfl_down(t3, off);
  }
  if (lane == 0) {
    tau_all[wrow * 3 + 0] = t0 + b_attn[0];
    tau_all[wrow * 3 + 1] = t1 + b_attn[1];
    tau_all[wrow * 3 + 2] = t2 + b_attn[2];
    tau_know[wrow]        = t3 + b_know[0];
  }
}

// ---------------------------------------------------------------- WMMA GEMM
// scores = Xb (R x D) @ Eb^T (N x D), fp32 out.
// Block = 256 thr (8 waves stacked in M); block tile = 128(M) x 64(N).
// B tile (64 cols x 64 K, bf16) staged into LDS via async global->LDS copies
// (double buffered, ASYNCcnt, ONE barrier per K-step), fed to WMMA through
// conflict-free ds_load_b128 (row stride 144 B). B fragments use distinct
// registers so DS loads clause and overlap the WMMAs.
template <int D>
__global__ void __launch_bounds__(256) gemm_bf16_kernel(const __bf16* __restrict__ X,
                                                        const __bf16* E,
                                                        float* __restrict__ out, int N) {
  constexpr int KSTEP = 64;            // K per staged buffer (2 WMMA sub-steps)
  constexpr int ROWB  = 144;           // bytes per LDS row: 64*2 data + 16 pad
  constexpr int BUFB  = 64 * ROWB;     // one buffer: 9216 B
  __shared__ __align__(16) char smem[2 * BUFB];

  const int tid  = threadIdx.x;
  const int lane = tid & 31;
  const int wave = tid >> 5;           // 0..7
  const int half = lane >> 4;          // 0: lanes 0-15, 1: lanes 16-31
  const int l15  = lane & 15;
  const int row0 = (blockIdx.y * 8 + wave) * 16;
  const int col0 = blockIdx.x * 64;

  // Async staging: thread -> (row sr, 32-byte quarter sq) of the 64x128B tile.
  // 256 threads x 2 async b128 = 8 KB = full tile.
  const int sr = tid & 63;
  const int sq = tid >> 6;             // 0..3
  const __bf16* gstage = E + (size_t)(col0 + sr) * D + sq * 16;
  const unsigned lbase = (unsigned)(size_t)(&smem[0]) + (unsigned)(sr * ROWB + sq * 32);

  v8f acc[4];
#pragma unroll
  for (int t = 0; t < 4; ++t) { v8f z = {}; acc[t] = z; }

  const __bf16* xrow = X + (size_t)(row0 + l15) * D;

#define STAGE(kt, buf)                                                      \
  do {                                                                      \
    const __bf16* g_ = gstage + (kt) * KSTEP;                               \
    unsigned l_ = lbase + (unsigned)((buf) * BUFB);                         \
    asm volatile(                                                           \
        "global_load_async_to_lds_b128 %0, %1, off\n"                      \
        "global_load_async_to_lds_b128 %0, %1, off offset:16"              \
        :: "v"(l_), "v"(g_) : "memory");                                    \
  } while (0)

  constexpr int NKT = D / KSTEP;       // 16
  STAGE(0, 0);
  for (int kt = 0; kt < NKT; ++kt) {
    // stage(kt) complete for this wave; barrier makes it complete for all
    // waves AND proves all waves finished compute(kt-1) (last readers of the
    // buffer stage(kt+1) will overwrite) -> single barrier per K-step.
    asm volatile("s_wait_asynccnt 0x0" ::: "memory");
    __syncthreads();
    if (kt + 1 < NKT) STAGE(kt + 1, (kt + 1) & 1);

    const char* buf = &smem[(kt & 1) * BUFB];
    const int k0 = kt * KSTEP;
#pragma unroll
    for (int s = 0; s < 2; ++s) {
      // A 16x32 bf16 fragment: half 0 -> K {0..7,16..23}; half 1 -> K {8..15,24..31}
      union { v16bf v; float4 f[2]; } a;
      a.f[0] = *(const float4*)(xrow + k0 + s * 32 + half * 8);
      a.f[1] = *(const float4*)(xrow + k0 + s * 32 + half * 8 + 16);
      // Preload all 4 B fragments into distinct registers (clauseable DS loads)
      union { v16bf v; float4 f[2]; } b[4];
#pragma unroll
      for (int t = 0; t < 4; ++t) {
        const char* brow = buf + (t * 16 + l15) * ROWB + s * 64 + half * 16;
        b[t].f[0] = *(const float4*)(brow);
        b[t].f[1] = *(const float4*)(brow + 32);
      }
#pragma unroll
      for (int t = 0; t < 4; ++t)
        acc[t] = __builtin_amdgcn_wmma_f32_16x16x32_bf16(
            false, a.v, false, b[t].v, (short)0, acc[t], false, false);
    }
  }
#undef STAGE

  // C/D layout: VGPR r, lanes 0-15 -> M=r; lanes 16-31 -> M=8+r; N = l15
#pragma unroll
  for (int t = 0; t < 4; ++t) {
    float* o = out + (size_t)(row0 + half * 8) * N + col0 + t * 16 + l15;
#pragma unroll
    for (int r = 0; r < 8; ++r) o[(size_t)r * N] = acc[t][r];
  }
}

// ---------------------------------------------------------------- gating
template <int PER, int NGATES>
__global__ void gate_kernel(const float* scores, const float* tau, int tau_stride,
                            float* out0, float* out1, float* acc0, float* acc1, int K) {
  constexpr int NT = 256;
  constexpr int N = PER * NT;
  const int row = blockIdx.x;
  const int tid = threadIdx.x;
  const int lane = tid & 31;
  __shared__ float sSum[NT];
  __shared__ float sMax[NT];
  __shared__ int sCnt;

  float sreg[PER];
  unsigned oreg[PER];
  const float* srow = scores + (size_t)row * N;
#pragma unroll
  for (int j = 0; j < PER; ++j) {
    float v = srow[tid + j * NT];
    sreg[j] = v;
    unsigned u = __float_as_uint(v);
    oreg[j] = (u & 0x80000000u) ? ~u : (u | 0x80000000u);  // monotone code
  }

  // kth = largest code t with count(code >= t) >= K  == K-th largest score
  unsigned kth = 0u;
  for (int bit = 31; bit >= 0; --bit) {
    unsigned trial = kth | (1u << bit);
    if (tid == 0) sCnt = 0;
    __syncthreads();
    int local = 0;
#pragma unroll
    for (int j = 0; j < PER; ++j) local += (oreg[j] >= trial) ? 1 : 0;
    for (int off = 16; off; off >>= 1) local += __shfl_down(local, off);
    if (lane == 0 && local) atomicAdd(&sCnt, local);
    __syncthreads();
    if (sCnt >= K) kth = trial;
    __syncthreads();
  }

  for (int g = 0; g < NGATES; ++g) {
    const float tv = tau[(size_t)row * tau_stride + g];
    float kreg[PER];
    float lsum = 0.f, lmax = 0.f;
#pragma unroll
    for (int j = 0; j < PER; ++j) {
      float raw  = sreg[j] - tv;
      float gate = (raw > 0.f) ? raw : 1e-8f * __expf(raw);
      float eg   = expm1f(gate);
      float kept = (oreg[j] >= kth) ? eg : 0.f;
      kreg[j] = kept;
      lsum += kept;
      lmax = fmaxf(lmax, kept);
    }
    sSum[tid] = lsum; sMax[tid] = lmax;
    __syncthreads();
    for (int off = NT / 2; off > 0; off >>= 1) {
      if (tid < off) {
        sSum[tid] += sSum[tid + off];
        sMax[tid] = fmaxf(sMax[tid], sMax[tid + off]);
      }
      __syncthreads();
    }
    const float scale = tanhf(sMax[0]) / (sSum[0] + 1e-8f);
    float* o = (g == 0) ? out0 : out1;
    float* a = (g == 0) ? acc0 : acc1;
#pragma unroll
    for (int j = 0; j < PER; ++j) {
      float v = kreg[j] * scale;
      o[(size_t)row * N + tid + j * NT] = v;
      if (v != 0.f) atomicAdd(&a[tid + j * NT], v);
    }
    __syncthreads();
  }
}

// ---------------------------------------------------------------- aux
__global__ void aux_kernel(const float* accQ, const float* accK, const float* accV,
                           const float* accKn, int Nqk, int Nv, int Nkn, float invR,
                           float* auxA, float* auxK) {
  __shared__ float s1[256], s2[256];
  const int tid = threadIdx.x;
  const float tqk = 1.f / (float)Nqk, tv = 1.f / (float)Nv, tkn = 1.f / (float)Nkn;
  float a1 = 0.f, a2 = 0.f;
  for (int i = tid; i < Nqk; i += 256) { float d = accQ[i] * invR - tqk; a1 += d * d * (float)Nqk; }
  for (int i = tid; i < Nqk; i += 256) { float d = accK[i] * invR - tqk; a1 += d * d * (float)Nqk; }
  for (int i = tid; i < Nv;  i += 256) { float d = accV[i] * invR - tv;  a1 += d * d * (float)Nv; }
  for (int i = tid; i < Nkn; i += 256) { float d = accKn[i] * invR - tkn; a2 += d * d * (float)Nkn; }
  s1[tid] = a1; s2[tid] = a2;
  __syncthreads();
  for (int off = 128; off > 0; off >>= 1) {
    if (tid < off) { s1[tid] += s1[tid + off]; s2[tid] += s2[tid + off]; }
    __syncthreads();
  }
  if (tid == 0) { *auxA = s1[0]; *auxK = s2[0]; }
}

// ---------------------------------------------------------------- launch
extern "C" void kernel_launch(void* const* d_in, const int* in_sizes, int n_in,
                              void* d_out, int out_size, void* d_ws, size_t ws_size,
                              hipStream_t stream) {
  const float* x      = (const float*)d_in[0];
  const float* qk_emb = (const float*)d_in[1];
  const float* v_emb  = (const float*)d_in[2];
  const float* kn_emb = (const float*)d_in[3];
  const float* w_attn = (const float*)d_in[4];
  const float* b_attn = (const float*)d_in[5];
  const float* w_know = (const float*)d_in[6];
  const float* b_know = (const float*)d_in[7];

  const int D   = in_sizes[4] / 3;   // 1024
  const int R   = in_sizes[0] / D;   // 4096 (= B*S)
  const int Nqk = in_sizes[1] / D;   // 4096
  const int Nv  = in_sizes[2] / D;   // 4096
  const int Nkn = in_sizes[3] / D;   // 8192

  float* out = (float*)d_out;
  float* gQ   = out;
  float* gK   = gQ + (size_t)R * Nqk;
  float* gV   = gK + (size_t)R * Nqk;
  float* auxA = gV + (size_t)R * Nv;
  float* gKn  = auxA + 1;
  float* auxK = gKn + (size_t)R * Nkn;

  char* ws = (char*)d_ws;
  __bf16* xb  = (__bf16*)ws;  ws += (size_t)R * D * 2;
  __bf16* qkb = (__bf16*)ws;  ws += (size_t)Nqk * D * 2;
  __bf16* vb  = (__bf16*)ws;  ws += (size_t)Nv * D * 2;
  __bf16* knb = (__bf16*)ws;  ws += (size_t)Nkn * D * 2;
  float* tau_all  = (float*)ws;  ws += (size_t)R * 3 * sizeof(float);
  float* tau_know = (float*)ws;  ws += (size_t)R * sizeof(float);
  float* acc = (float*)ws;       // 3*Nqk + Nkn floats, contiguous
  float* accQ = acc, *accK = acc + Nqk, *accV = acc + 2 * Nqk, *accKn = acc + 3 * Nqk;

  // 1) fp32 -> bf16
  cvt_bf16_kernel<<<(R * D + 255) / 256, 256, 0, stream>>>(x, xb, R * D);
  cvt_bf16_kernel<<<(Nqk * D + 255) / 256, 256, 0, stream>>>(qk_emb, qkb, Nqk * D);
  cvt_bf16_kernel<<<(Nv * D + 255) / 256, 256, 0, stream>>>(v_emb, vb, Nv * D);
  cvt_bf16_kernel<<<(Nkn * D + 255) / 256, 256, 0, stream>>>(kn_emb, knb, Nkn * D);

  // 2) zero column accumulators
  const int nAcc = 3 * Nqk + Nkn;
  zero_f32_kernel<<<(nAcc + 255) / 256, 256, 0, stream>>>(acc, nAcc);

  // 3) taus (wave per row)
  tau_kernel<<<(R * 32 + 255) / 256, 256, 0, stream>>>(x, w_attn, b_attn, w_know, b_know,
                                                       tau_all, tau_know, R, D);

  // 4) WMMA score GEMMs (async-LDS staged B), scores staged in-place in gate outputs
  gemm_bf16_kernel<1024><<<dim3(Nqk / 64, R / 128), 256, 0, stream>>>(xb, qkb, gQ, Nqk);
  gemm_bf16_kernel<1024><<<dim3(Nv / 64, R / 128), 256, 0, stream>>>(xb, vb, gV, Nv);
  gemm_bf16_kernel<1024><<<dim3(Nkn / 64, R / 128), 256, 0, stream>>>(xb, knb, gKn, Nkn);

  // 5) gates (block per row); Q & K share the same top-k set of scores_qk
  gate_kernel<16, 2><<<R, 256, 0, stream>>>(gQ, tau_all, 3, gQ, gK, accQ, accK, 64);
  gate_kernel<16, 1><<<R, 256, 0, stream>>>(gV, tau_all + 2, 3, gV, nullptr, accV, nullptr, 64);
  gate_kernel<32, 1><<<R, 256, 0, stream>>>(gKn, tau_know, 1, gKn, nullptr, accKn, nullptr, 128);

  // 6) aux scalars
  aux_kernel<<<1, 256, 0, stream>>>(accQ, accK, accV, accKn, Nqk, Nv, Nkn, 1.0f / (float)R,
                                    auxA, auxK);
}
